// GINNet_8993661518249
// MI455X (gfx1250) — matrix operand order repaired and checked by reading, not compiled, not run on hardware
//
#include <hip/hip_runtime.h>

#define HC 96  // hidden width, fixed by the model

typedef __attribute__((ext_vector_type(16))) __bf16 bf16x16;
typedef __attribute__((ext_vector_type(8)))  __bf16 bf16x8;
typedef __attribute__((ext_vector_type(8)))  float  floatx8;

__device__ __forceinline__ __bf16 f2bf(float f) {
  unsigned u = __builtin_bit_cast(unsigned, f);
  u += 0x7fffu + ((u >> 16) & 1u);           // round-to-nearest-even
  unsigned short h = (unsigned short)(u >> 16);
  return __builtin_bit_cast(__bf16, h);
}

// ---------------- per-column batch-norm statistics ----------------
__global__ void col_stats_kernel(const float* __restrict__ A, int rows, int cols,
                                 float* __restrict__ mu, float* __restrict__ rsig) {
  __shared__ float ssum[256], ssq[256];
  const int c = blockIdx.x;
  float s = 0.f, s2 = 0.f;
  for (int r = threadIdx.x; r < rows; r += blockDim.x) {
    float v = A[(size_t)r * cols + c];
    s += v; s2 += v * v;
  }
  ssum[threadIdx.x] = s; ssq[threadIdx.x] = s2;
  __syncthreads();
  for (int off = 128; off > 0; off >>= 1) {
    if ((int)threadIdx.x < off) {
      ssum[threadIdx.x] += ssum[threadIdx.x + off];
      ssq[threadIdx.x]  += ssq[threadIdx.x + off];
    }
    __syncthreads();
  }
  if (threadIdx.x == 0) {
    float m   = ssum[0] / rows;
    float var = ssq[0] / rows - m * m;
    mu[c]   = m;
    rsig[c] = rsqrtf(var + 1e-5f);
  }
}

// ---------------- BN (+relu) then convert to bf16 ----------------
__global__ void bn_cvt_bf16_kernel(const float* __restrict__ A,
                                   const float* __restrict__ mu, const float* __restrict__ rsig,
                                   const float* __restrict__ w,  const float* __restrict__ b,
                                   __bf16* __restrict__ out, int n, int cols, int relu) {
  int i = blockIdx.x * blockDim.x + threadIdx.x;
  if (i >= n) return;
  int j = i % cols;
  float v = (A[i] - mu[j]) * rsig[j] * w[j] + b[j];
  if (relu) v = fmaxf(v, 0.f);
  out[i] = f2bf(v);
}

__global__ void f32_to_bf16_kernel(const float* __restrict__ in, __bf16* __restrict__ out, int n) {
  int i = blockIdx.x * blockDim.x + threadIdx.x;
  if (i < n) out[i] = f2bf(in[i]);
}

__global__ void copy_f32_kernel(const float* __restrict__ in, float* __restrict__ out, int n) {
  int i = blockIdx.x * blockDim.x + threadIdx.x;
  if (i < n) out[i] = in[i];
}

__global__ void zero_f32_kernel(float* __restrict__ p, int n) {
  int i = blockIdx.x * blockDim.x + threadIdx.x;
  if (i < n) p[i] = 0.f;
}

// ---------------- pack f32 weight [K,96] into WMMA B-fragment layout (bf16) ------
// Output element index: (((s*6 + t)*32 + L)*16 + e)
//   s = k-step (32 K per step), t = 16-col tile, L = lane, e = fragment element.
// Lane L holds K = s*32 + (e<8 ? (L/16)*8 + e : 16 + (L/16)*8 + (e-8)) of column t*16 + (L%16), per ISA 7.12.2.
__global__ void pack_b_kernel(const float* __restrict__ W, __bf16* __restrict__ Bp, int K) {
  int i = blockIdx.x * blockDim.x + threadIdx.x;
  int total = K * HC;
  if (i >= total) return;
  int e = i & 15;
  int L = (i >> 4) & 31;
  int t = (i >> 9) % 6;
  int s = i / (16 * 32 * 6);
  int k   = s * 32 + (e < 8 ? (L >> 4) * 8 + e : 16 + (L >> 4) * 8 + (e - 8));
  int col = t * 16 + (L & 15);
  Bp[i] = f2bf(W[(size_t)k * HC + col]);
}

// ---------------- GIN neighbor sum: agg[dst] += h[src] ----------------
__global__ void scatter_add_kernel(const float* __restrict__ h, const int* __restrict__ src,
                                   const int* __restrict__ dst, float* __restrict__ agg, int E) {
  int i = blockIdx.x * blockDim.x + threadIdx.x;
  int total = E * HC;
  if (i >= total) return;
  int e = i / HC;
  int f = i - e * HC;
  atomicAdd(&agg[(size_t)dst[e] * HC + f], h[(size_t)src[e] * HC + f]);
}

// ---------------- global_add_pool ----------------
__global__ void pool_add_kernel(const float* __restrict__ h, const int* __restrict__ batch,
                                float* __restrict__ g, int N) {
  int i = blockIdx.x * blockDim.x + threadIdx.x;
  int total = N * HC;
  if (i >= total) return;
  int n = i / HC;
  int f = i - n * HC;
  atomicAdd(&g[(size_t)batch[n] * HC + f], h[i]);
}

// ---------------- WMMA GEMM: C[M,96] = relu?(A[M,K]_bf16 @ B[K,96] + bias) -------
// 192 threads = 6 waves/block; wave w owns 16 columns [16w,16w+16) and TWO 16-row
// tiles (32 rows per block) sharing one packed-B fragment per k-step.
// The second tile's A pointer is clamped into bounds so the k-loop is branch-free
// (EXEC stays all-ones for WMMA); only the tile-2 store is guarded.
__global__ __launch_bounds__(192) void gemm_bf16_wmma_kernel(
    const __bf16* __restrict__ A, const __bf16* __restrict__ Bpack,
    const float* __restrict__ bias, float* __restrict__ C,
    int M, int K, int relu) {
  const int lane = threadIdx.x & 31;
  const int wave = threadIdx.x >> 5;     // 0..5 -> column tile
  const int g    = lane >> 4;            // lane group (0/1)
  const int lr   = lane & 15;            // lane within group
  const int row0 = blockIdx.x << 5;      // 32 rows per block
  const int col0 = wave << 4;
  if (row0 >= M) return;
  const bool has2 = (row0 + 16) < M;     // M is a multiple of 16

  const int r1 = min(row0 + 16 + lr, M - 1);   // clamped: in-bounds dummy reads
  const __bf16* Arow0 = A + (size_t)(row0 + lr) * K;
  const __bf16* Arow1 = A + (size_t)r1 * K;
  const __bf16* Bp    = Bpack + ((size_t)wave * 32 + lane) * 16;  // + s*3072 per step

  floatx8 acc0 = {}, acc1 = {};

  for (int k0 = 0, s = 0; k0 < K; k0 += 32, ++s) {
    bf16x16 bm = *reinterpret_cast<const bf16x16*>(Bp + (size_t)s * (6 * 32 * 16));

    bf16x8 alo = *reinterpret_cast<const bf16x8*>(Arow0 + k0 + g * 8);
    bf16x8 ahi = *reinterpret_cast<const bf16x8*>(Arow0 + k0 + 16 + g * 8);
    bf16x16 a0;
#pragma unroll
    for (int i = 0; i < 8; ++i) { a0[i] = alo[i]; a0[8 + i] = ahi[i]; }
    acc0 = __builtin_amdgcn_wmma_f32_16x16x32_bf16(false, a0, false, bm,
                                                   (short)0, acc0, false, false);

    bf16x8 blo = *reinterpret_cast<const bf16x8*>(Arow1 + k0 + g * 8);
    bf16x8 bhi = *reinterpret_cast<const bf16x8*>(Arow1 + k0 + 16 + g * 8);
    bf16x16 a1;
#pragma unroll
    for (int i = 0; i < 8; ++i) { a1[i] = blo[i]; a1[8 + i] = bhi[i]; }
    acc1 = __builtin_amdgcn_wmma_f32_16x16x32_bf16(false, a1, false, bm,
                                                   (short)0, acc1, false, false);
  }

  const int   col = col0 + lr;
  const float bv  = bias[col];
#pragma unroll
  for (int r = 0; r < 8; ++r) {
    int   row = row0 + g * 8 + r;        // C/D: VGPR r -> M = r + 8*laneGroup
    float v   = acc0[r] + bv;
    if (relu) v = fmaxf(v, 0.f);
    C[(size_t)row * HC + col] = v;
  }
  if (has2) {                             // wave-uniform store guard only
#pragma unroll
    for (int r = 0; r < 8; ++r) {
      int   row = row0 + 16 + g * 8 + r;
      float v   = acc1[r] + bv;
      if (relu) v = fmaxf(v, 0.f);
      C[(size_t)row * HC + col] = v;
    }
  }
}

// ---------------- final BN + 96x10 classifier + log_softmax ----------------
__global__ void classifier_kernel(const float* __restrict__ g2,
                                  const float* __restrict__ mu, const float* __restrict__ rsig,
                                  const float* __restrict__ bw, const float* __restrict__ bb,
                                  const float* __restrict__ Wc, const float* __restrict__ bc,
                                  float* __restrict__ out, int G) {
  int r = blockIdx.x * blockDim.x + threadIdx.x;
  if (r >= G) return;
  float logit[10];
#pragma unroll
  for (int c = 0; c < 10; ++c) logit[c] = bc[c];
  for (int k = 0; k < HC; ++k) {
    float v = (g2[(size_t)r * HC + k] - mu[k]) * rsig[k] * bw[k] + bb[k];
#pragma unroll
    for (int c = 0; c < 10; ++c) logit[c] += v * Wc[k * 10 + c];
  }
  float mx = logit[0];
#pragma unroll
  for (int c = 1; c < 10; ++c) mx = fmaxf(mx, logit[c]);
  float sum = 0.f;
#pragma unroll
  for (int c = 0; c < 10; ++c) sum += expf(logit[c] - mx);
  float lse = mx + logf(sum);
#pragma unroll
  for (int c = 0; c < 10; ++c) out[(size_t)r * 10 + c] = logit[c] - lse;
}

#define CEILDIV(a, b) (((a) + (b) - 1) / (b))

extern "C" void kernel_launch(void* const* d_in, const int* in_sizes, int n_in,
                              void* d_out, int out_size, void* d_ws, size_t ws_size,
                              hipStream_t stream) {
  const float* x         = (const float*)d_in[0];
  const int*   ei        = (const int*)d_in[1];
  const int*   batch     = (const int*)d_in[2];
  const float* bn_feat_w = (const float*)d_in[3];
  const float* bn_feat_b = (const float*)d_in[4];
  const float* Wf        = (const float*)d_in[5];
  const float* bf        = (const float*)d_in[6];
  const float* bn_fc_w   = (const float*)d_in[7];
  const float* bn_fc_b   = (const float*)d_in[8];
  const float* Wl        = (const float*)d_in[9];
  const float* bl        = (const float*)d_in[10];
  const float* bn_hid_w  = (const float*)d_in[11];
  const float* bn_hid_b  = (const float*)d_in[12];
  const float* Wc        = (const float*)d_in[13];
  const float* bc        = (const float*)d_in[14];
  const float *W1[3], *b1[3], *bnw[3], *bnb[3], *W2[3], *b2[3];
  for (int i = 0; i < 3; ++i) {
    W1[i]  = (const float*)d_in[15 + 6 * i];
    b1[i]  = (const float*)d_in[16 + 6 * i];
    bnw[i] = (const float*)d_in[17 + 6 * i];
    bnb[i] = (const float*)d_in[18 + 6 * i];
    W2[i]  = (const float*)d_in[19 + 6 * i];
    b2[i]  = (const float*)d_in[20 + 6 * i];
  }

  const int N   = in_sizes[2];          // 50000
  const int E   = in_sizes[1] / 2;      // 800000
  const int FIN = in_sizes[3];          // 128
  const int G   = out_size / 10;        // 512
  const int* src = ei;
  const int* dst = ei + E;

  // ---- workspace layout (256B aligned slots) ----
  char* ws = (char*)d_ws;
  size_t off = 0;
  auto take = [&](size_t bytes) -> char* {
    char* p = ws + off;
    off = (off + bytes + 255) & ~(size_t)255;
    return p;
  };
  float*  h    = (float*)take((size_t)N * HC * 4);     // node features (f32)
  float*  z    = (float*)take((size_t)N * HC * 4);     // agg, then pre-BN linear output
  __bf16* abf  = (__bf16*)take((size_t)N * FIN * 2);   // bf16 staging: xbf / ubf / mbf
  __bf16* Wfb  = (__bf16*)take((size_t)FIN * HC * 2);  // packed-B buffers
  __bf16* W1b[3]; __bf16* W2b[3];
  for (int i = 0; i < 3; ++i) {
    W1b[i] = (__bf16*)take((size_t)HC * HC * 2);
    W2b[i] = (__bf16*)take((size_t)HC * HC * 2);
  }
  __bf16* Wlb  = (__bf16*)take((size_t)HC * HC * 2);
  float*  stats = (float*)take(1024 * 4);
  float*  gp   = (float*)take((size_t)G * HC * 4);     // pooled
  float*  g2   = (float*)take((size_t)G * HC * 4);     // after first pooled linear
  float* mu  = stats;        float* rsig  = stats + 256;
  float* mu2 = stats + 512;  float* rsig2 = stats + 768;

  // ---- weight conversion + packing into WMMA B-fragment layout ----
  pack_b_kernel<<<CEILDIV(FIN * HC, 256), 256, 0, stream>>>(Wf, Wfb, FIN);
  for (int i = 0; i < 3; ++i) {
    pack_b_kernel<<<CEILDIV(HC * HC, 256), 256, 0, stream>>>(W1[i], W1b[i], HC);
    pack_b_kernel<<<CEILDIV(HC * HC, 256), 256, 0, stream>>>(W2[i], W2b[i], HC);
  }
  pack_b_kernel<<<CEILDIV(HC * HC, 256), 256, 0, stream>>>(Wl, Wlb, HC);

  // ---- input BN + first linear: h = relu(BN(x) @ Wf + bf) ----
  col_stats_kernel<<<FIN, 256, 0, stream>>>(x, N, FIN, mu, rsig);
  bn_cvt_bf16_kernel<<<CEILDIV(N * FIN, 256), 256, 0, stream>>>(
      x, mu, rsig, bn_feat_w, bn_feat_b, abf, N * FIN, FIN, 0);
  gemm_bf16_wmma_kernel<<<CEILDIV(N, 32), 192, 0, stream>>>(abf, Wfb, bf, h, N, FIN, 1);

  // ---- 3 GIN layers ----
  for (int i = 0; i < 3; ++i) {
    copy_f32_kernel<<<CEILDIV(N * HC, 256), 256, 0, stream>>>(h, z, N * HC);          // (1+eps)*h, eps=0
    scatter_add_kernel<<<CEILDIV(E * HC, 256), 256, 0, stream>>>(h, src, dst, z, E);  // + sum_neighbors
    f32_to_bf16_kernel<<<CEILDIV(N * HC, 256), 256, 0, stream>>>(z, abf, N * HC);
    gemm_bf16_wmma_kernel<<<CEILDIV(N, 32), 192, 0, stream>>>(abf, W1b[i], b1[i], z, N, HC, 0);
    col_stats_kernel<<<HC, 256, 0, stream>>>(z, N, HC, mu, rsig);
    bn_cvt_bf16_kernel<<<CEILDIV(N * HC, 256), 256, 0, stream>>>(
        z, mu, rsig, bnw[i], bnb[i], abf, N * HC, HC, 1);
    gemm_bf16_wmma_kernel<<<CEILDIV(N, 32), 192, 0, stream>>>(abf, W2b[i], b2[i], h, N, HC, 1);
  }

  // ---- pooled head ----
  zero_f32_kernel<<<CEILDIV(G * HC, 256), 256, 0, stream>>>(gp, G * HC);
  pool_add_kernel<<<CEILDIV(N * HC, 256), 256, 0, stream>>>(h, batch, gp, N);
  col_stats_kernel<<<HC, 256, 0, stream>>>(gp, G, HC, mu, rsig);
  bn_cvt_bf16_kernel<<<CEILDIV(G * HC, 256), 256, 0, stream>>>(
      gp, mu, rsig, bn_fc_w, bn_fc_b, abf, G * HC, HC, 0);
  gemm_bf16_wmma_kernel<<<CEILDIV(G, 32), 192, 0, stream>>>(abf, Wlb, bl, g2, G, HC, 1);

  // ---- final BN + classifier + log_softmax ----
  col_stats_kernel<<<HC, 256, 0, stream>>>(g2, G, HC, mu2, rsig2);
  classifier_kernel<<<CEILDIV(G, 128), 128, 0, stream>>>(
      g2, mu2, rsig2, bn_hid_w, bn_hid_b, Wc, bc, (float*)d_out, G);
}
